// Attention_72404558676065
// MI455X (gfx1250) — matrix operand order repaired and checked by reading, not compiled, not run on hardware
//
#include <hip/hip_runtime.h>

typedef _Float16 v2h  __attribute__((ext_vector_type(2)));
typedef _Float16 v8h  __attribute__((ext_vector_type(8)));
typedef _Float16 v16h __attribute__((ext_vector_type(16)));
typedef float    v8f  __attribute__((ext_vector_type(8)));

#define B_   8
#define C_   256
#define N_   2048
#define MT   128        // query rows per block (16 per wave, 8 waves)
#define NT   32         // key/value columns per j-step
#define QP   136        // qbuf pitch in halves (128 + 8 pad)
#define KJCP 264        // kjc pitch in halves (256 + 8 pad)
#define KCJP 40         // kcj pitch in halves (32 + 8 pad)
#define PBP  40         // pbuf pitch in halves

union SLDS {
    _Float16 qbuf[MT * QP];                 // 34816 B (Q staging, two passes)
    struct {
        _Float16 kjc[NT * KJCP];            // 16896 B  K tile as [j][c]
        _Float16 kcj[C_ * KCJP];            // 20480 B  K tile as [c][j]
        _Float16 pbuf[8 * 16 * PBP];        // 10240 B  per-wave P strips
    } loop;                                 // total 47616 B
};

union F16x16 { v16h v; v8h h[2]; };

// CDNA5 async global->LDS copy, 16B per lane. LDS dest = addr[31:0] of the
// generic shared pointer (HW takes VGPR[VDST] as LDS byte offset).
__device__ __forceinline__ void async_copy_b128(const _Float16* gp,
                                                _Float16* lp) {
    unsigned           lds_off = (unsigned)(unsigned long long)lp;
    unsigned long long gaddr   = (unsigned long long)gp;
    asm volatile("global_load_async_to_lds_b128 %0, %1, off"
                 :: "v"(lds_off), "v"(gaddr)
                 : "memory");
}

__device__ __forceinline__ void wait_async_zero() {
#if __has_builtin(__builtin_amdgcn_s_wait_asynccnt)
    __builtin_amdgcn_s_wait_asynccnt(0);
#else
    asm volatile("s_wait_asynccnt 0" ::: "memory");
#endif
}

// ---------------- Pass 1: convert x (f32) -> xh (f16) once ----------------
__global__ void convert_f32_f16_kernel(const float* __restrict__ x,
                                       _Float16* __restrict__ xh) {
    size_t i = ((size_t)blockIdx.x * 256 + threadIdx.x) * 8;
    float4 a = ((const float4*)(x + i))[0];
    float4 b = ((const float4*)(x + i))[1];
    v8h h = { (_Float16)a.x, (_Float16)a.y, (_Float16)a.z, (_Float16)a.w,
              (_Float16)b.x, (_Float16)b.y, (_Float16)b.z, (_Float16)b.w };
    *(v8h*)(xh + i) = h;
}

// ---------------- Pass 2: fused flash attention ----------------
__launch_bounds__(256, 1)
__global__ void attn_flash_kernel(const _Float16* __restrict__ xh,
                                  float* __restrict__ out) {
    __shared__ __align__(16) SLDS s;

    const int tid  = threadIdx.x;
    const int lane = tid & 31;
    const int wave = tid >> 5;
    const int g    = lane >> 4;      // lane half-group (0/1)
    const int lr   = lane & 15;      // lane within group
    const int b    = blockIdx.y;
    const int i0   = blockIdx.x * MT;
    const _Float16* __restrict__ xb = xh + (size_t)b * C_ * N_;

    // ---- Stage Q, build A-fragments in registers (two channel halves) ----
    v16h qa[8];                       // 16 rows x 256 channels of Q (f16)
    for (int p = 0; p < 2; ++p) {
        __syncthreads();
        {
            const int cl = tid >> 1;          // local channel row 0..127
            const int h  = tid & 1;           // i half
            const _Float16* src = xb + (size_t)(p * 128 + cl) * N_ + i0 + h * 64;
            #pragma unroll
            for (int q = 0; q < 8; ++q) {
                v8h v = *(const v8h*)(src + q * 8);
                #pragma unroll
                for (int e = 0; e < 8; ++e)
                    s.qbuf[(size_t)(h * 64 + q * 8 + e) * QP + cl] = v[e];
            }
        }
        __syncthreads();
        const _Float16* qrow = s.qbuf + (size_t)(wave * 16 + lr) * QP;
        #pragma unroll
        for (int k = 0; k < 4; ++k) {
            int c0 = k * 32;
            F16x16 u;
            u.h[0] = *(const v8h*)(qrow + c0 + 16 * g);       // K = 16g..
            u.h[1] = *(const v8h*)(qrow + c0 + 16 * g + 8);   // K = 16g+8..
            qa[p * 4 + k] = u.v;
        }
    }

    // ---- Flash-attention state ----
    v8f o[16];
    #pragma unroll
    for (int t = 0; t < 16; ++t)
        #pragma unroll
        for (int e = 0; e < 8; ++e) o[t][e] = 0.0f;
    float mrow[8], lrow[8];
    #pragma unroll
    for (int r = 0; r < 8; ++r) { mrow[r] = -__builtin_inff(); lrow[r] = 0.0f; }

    // ---- Main j loop ----
    for (int jt = 0; jt < N_ / NT; ++jt) {
        const int j0 = jt * NT;
        __syncthreads();   // previous step's LDS reads complete
        {
            // kcj [c][j]: raw f16 copy via async-to-LDS (16B per lane, x4)
            #pragma unroll
            for (int q = 0; q < 4; ++q) {
                int idx = q * 256 + tid;
                int c   = idx >> 2;
                int seg = idx & 3;
                async_copy_b128(xb + (size_t)c * N_ + j0 + seg * 8,
                                s.loop.kcj + (size_t)c * KCJP + seg * 8);
            }
            // kjc [j][c]: transpose, thread owns a c-row pair -> b32 stores
            const int c0 = (tid & 127) * 2;
            const int h  = tid >> 7;
            const _Float16* ra = xb + (size_t)c0 * N_ + j0 + h * 16;
            const _Float16* rb = ra + N_;
            v8h a0 = *(const v8h*)(ra);
            v8h a1 = *(const v8h*)(ra + 8);
            v8h b0 = *(const v8h*)(rb);
            v8h b1 = *(const v8h*)(rb + 8);
            #pragma unroll
            for (int jj = 0; jj < 8; ++jj) {
                v2h p0 = { a0[jj], b0[jj] };
                v2h p1 = { a1[jj], b1[jj] };
                *(v2h*)(s.loop.kjc + (size_t)(h * 16 + jj)     * KJCP + c0) = p0;
                *(v2h*)(s.loop.kjc + (size_t)(h * 16 + 8 + jj) * KJCP + c0) = p1;
            }
            if (jt + 1 < N_ / NT)
                __builtin_prefetch(xb + (size_t)tid * N_ + j0 + NT, 0, 1);
        }
        wait_async_zero();   // our async transfers landed before the barrier
        __syncthreads();

        // ---- S = Q * K^T : k outer so consecutive WMMAs share A (reuse_a) ----
        v8f sacc[2];
        #pragma unroll
        for (int t = 0; t < 2; ++t)
            #pragma unroll
            for (int e = 0; e < 8; ++e) sacc[t][e] = 0.0f;
        #pragma unroll
        for (int k = 0; k < 8; ++k) {
            const _Float16* krow0 =
                s.loop.kjc + (size_t)(0 * 16 + lr) * KJCP + 32 * k + 16 * g;
            const _Float16* krow1 =
                s.loop.kjc + (size_t)(1 * 16 + lr) * KJCP + 32 * k + 16 * g;
            F16x16 u0, u1;
            u0.h[0] = *(const v8h*)(krow0);
            u0.h[1] = *(const v8h*)(krow0 + 8);
            u1.h[0] = *(const v8h*)(krow1);
            u1.h[1] = *(const v8h*)(krow1 + 8);
            sacc[0] = __builtin_amdgcn_wmma_f32_16x16x32_f16(
                false, qa[k], false, u0.v, (short)0, sacc[0], false, false);
            sacc[1] = __builtin_amdgcn_wmma_f32_16x16x32_f16(
                false, qa[k], false, u1.v, (short)0, sacc[1], true, false);
        }

        // ---- online softmax (row r+8g lives at VGPR index r) ----
        float mx[8];
        #pragma unroll
        for (int r = 0; r < 8; ++r) mx[r] = fmaxf(sacc[0][r], sacc[1][r]);
        #pragma unroll
        for (int d = 1; d <= 8; d <<= 1)
            #pragma unroll
            for (int r = 0; r < 8; ++r)
                mx[r] = fmaxf(mx[r], __shfl_xor(mx[r], d, 32));
        float scale[8];
        #pragma unroll
        for (int r = 0; r < 8; ++r) {
            float mn = fmaxf(mrow[r], mx[r]);
            scale[r] = __expf(mrow[r] - mn);
            mrow[r] = mn;
        }
        #pragma unroll
        for (int t = 0; t < 2; ++t)
            #pragma unroll
            for (int r = 0; r < 8; ++r)
                sacc[t][r] = __expf(sacc[t][r] - mrow[r]);
        float rs[8];
        #pragma unroll
        for (int r = 0; r < 8; ++r) rs[r] = sacc[0][r] + sacc[1][r];
        #pragma unroll
        for (int d = 1; d <= 8; d <<= 1)
            #pragma unroll
            for (int r = 0; r < 8; ++r)
                rs[r] += __shfl_xor(rs[r], d, 32);
        #pragma unroll
        for (int r = 0; r < 8; ++r)
            lrow[r] = lrow[r] * scale[r] + rs[r];

        float smin = scale[0];
        #pragma unroll
        for (int r = 1; r < 8; ++r) smin = fminf(smin, scale[r]);
        if (!__all(smin == 1.0f)) {
            #pragma unroll
            for (int t = 0; t < 16; ++t)
                #pragma unroll
                for (int r = 0; r < 8; ++r) o[t][r] *= scale[r];
        }

        // ---- write P (f16) to wave-private LDS strip ----
        _Float16* pw = s.loop.pbuf + (size_t)wave * 16 * PBP;
        #pragma unroll
        for (int t = 0; t < 2; ++t)
            #pragma unroll
            for (int r = 0; r < 8; ++r)
                pw[(size_t)(r + 8 * g) * PBP + t * 16 + lr] =
                    (_Float16)sacc[t][r];

        // ---- P A-fragment (16x32) ----
        const _Float16* prow = pw + (size_t)lr * PBP;
        F16x16 pu;
        pu.h[0] = *(const v8h*)(prow + 16 * g);
        pu.h[1] = *(const v8h*)(prow + 16 * g + 8);

        // ---- O += P * V : same A across all 16 tiles (reuse_a) ----
        {
            const _Float16* vrow = s.loop.kcj + (size_t)lr * KCJP;
            F16x16 vu;
            vu.h[0] = *(const v8h*)(vrow + 16 * g);
            vu.h[1] = *(const v8h*)(vrow + 16 * g + 8);
            o[0] = __builtin_amdgcn_wmma_f32_16x16x32_f16(
                false, pu.v, false, vu.v, (short)0, o[0], false, false);
        }
        #pragma unroll
        for (int t = 1; t < 16; ++t) {
            const _Float16* vrow = s.loop.kcj + (size_t)(t * 16 + lr) * KCJP;
            F16x16 vu;
            vu.h[0] = *(const v8h*)(vrow + 16 * g);
            vu.h[1] = *(const v8h*)(vrow + 16 * g + 8);
            o[t] = __builtin_amdgcn_wmma_f32_16x16x32_f16(
                false, pu.v, false, vu.v, (short)0, o[t], true, false);
        }
    }

    // ---- Epilogue: normalize + transposed store through LDS ----
    __syncthreads();
    float rinv[8];
    #pragma unroll
    for (int r = 0; r < 8; ++r) rinv[r] = 1.0f / lrow[r];

    float* scratch = (float*)&s + (size_t)wave * 544;   // 2176 B per wave
    const int iw = i0 + wave * 16;
    for (int tp = 0; tp < 8; ++tp) {
        #pragma unroll
        for (int tt = 0; tt < 2; ++tt) {
            int t = tp * 2 + tt;
            #pragma unroll
            for (int r = 0; r < 8; ++r)
                scratch[(size_t)(tt * 16 + lr) * 17 + (r + 8 * g)] =
                    o[t][r] * rinv[r];
        }
        #pragma unroll
        for (int q = 0; q < 16; ++q) {
            int idx = q * 32 + lane;
            int cl  = idx >> 4;
            int il  = idx & 15;
            float v = scratch[(size_t)cl * 17 + il];
            int cg  = tp * 32 + cl;
            out[(size_t)b * C_ * N_ + (size_t)cg * N_ + iw + il] = v;
        }
    }
}

extern "C" void kernel_launch(void* const* d_in, const int* in_sizes, int n_in,
                              void* d_out, int out_size, void* d_ws, size_t ws_size,
                              hipStream_t stream) {
    (void)in_sizes; (void)n_in; (void)out_size; (void)ws_size;
    const float* x  = (const float*)d_in[0];
    _Float16*   xh  = (_Float16*)d_ws;          // needs B*C*N*2 = 8.4 MB scratch
    float*      out = (float*)d_out;

    convert_f32_f16_kernel<<<dim3((B_ * C_ * N_) / (256 * 8)), dim3(256), 0,
                             stream>>>(x, xh);
    dim3 grid(N_ / MT, B_);
    attn_flash_kernel<<<grid, dim3(256), 0, stream>>>(xh, out);
}
